// MockLlamaAttention_4080218931649
// MI455X (gfx1250) — matrix operand order repaired and checked by reading, compiled, not run on hardware
//
#include <hip/hip_runtime.h>
#include <hip/hip_bf16.h>

// ---------------------------------------------------------------------------
// MockLlamaAttention for MI455X (gfx1250).
//   Stage 0: X fp32 -> bf16; W* fp32 -> bf16 transposed (Wt[n][k])
//   Stage 1: Q/K = Xb @ Wt (TDM-staged bf16 WMMA GEMM), V written head-transposed
//   Stage 2: flash attention (TDM-staged K/Vt tiles, WMMA QK^T + PV, online softmax)
//   Stage 3: out = Ob @ Wot  (f32 output)
// All LDS staging via tensor_load_to_lds (TENSORcnt), double-buffered.
// B=2 S=2048 D=2048 H=16 HD=128
// ---------------------------------------------------------------------------

#define B_  2
#define S_  2048
#define D_  2048
#define H_  16
#define HD_ 128

typedef __attribute__((ext_vector_type(16))) __bf16 v16bf;
typedef __attribute__((ext_vector_type(8)))  __bf16 v8bf;
typedef __attribute__((ext_vector_type(8)))  float  v8f;
typedef unsigned int v4u  __attribute__((ext_vector_type(4)));
typedef int          v8i_ __attribute__((ext_vector_type(8)));
typedef int          v4i_ __attribute__((ext_vector_type(4)));

__device__ __forceinline__ __bf16 f2bf(float f) {
  union { float f; unsigned u; } a; a.f = f;
  unsigned r = a.u + 0x7FFFu + ((a.u >> 16) & 1u);     // round-to-nearest-even
  union { unsigned short s; __bf16 b; } o; o.s = (unsigned short)(r >> 16);
  return o.b;
}
__device__ __forceinline__ unsigned pack2(float a, float b) {
  union { __bf16 b; unsigned short s; } x, y; x.b = f2bf(a); y.b = f2bf(b);
  return (unsigned)x.s | ((unsigned)y.s << 16);
}
__device__ __forceinline__ unsigned short bf2u(float f) {
  union { __bf16 b; unsigned short s; } u; u.b = f2bf(f); return u.s;
}
__device__ __forceinline__ unsigned lds_off(const void* p) {
  return (unsigned)(size_t)p;     // LDS aperture: low 32 bits = LDS byte offset
}

// ---------------------------------------------------------------------------
// TDM: 2-D tile (bf16 elements, dim0 contiguous) from global -> LDS.
// D# per CDNA5 ISA 8.3/8.4: group0 {count=1, lds_addr, global_addr, type=2},
// group1 {data_size=1(2B), tensor dims, tile dims, 48-bit dim0 stride}.
// Issue from one wave only; EXEC is ignored by TDM.
// 6-arg builtin (clang-23 / therock): (v4u, v8i, v4i, v4i, v8i, i32 cpol).
// ---------------------------------------------------------------------------
__device__ __forceinline__ void tdm_load_2d(unsigned ldsaddr, const void* gptr,
                                            unsigned tile_d0, unsigned tile_d1,
                                            unsigned tensor_d0, unsigned tensor_d1,
                                            unsigned long long stride0_elems) {
  unsigned long long ga = (unsigned long long)(size_t)gptr;
  v4u g0;
  g0[0] = 1u;                                            // count=1 valid descriptor
  g0[1] = ldsaddr;                                       // lds_addr (bytes)
  g0[2] = (unsigned)(ga & 0xFFFFFFFFu);                  // global_addr[31:0]
  g0[3] = (unsigned)((ga >> 32) & 0x01FFFFFFu) | (2u << 30);  // addr[56:32] | type=2
  v8i_ g1;
  g1[0] = (int)(1u << 16);                               // data_size=1 -> 2 bytes
  g1[1] = (int)((tensor_d0 & 0xFFFFu) << 16);            // tensor_dim0[15:0]
  g1[2] = (int)(((tensor_d0 >> 16) & 0xFFFFu) | ((tensor_d1 & 0xFFFFu) << 16));
  g1[3] = (int)(((tensor_d1 >> 16) & 0xFFFFu) | ((tile_d0 & 0xFFFFu) << 16));
  g1[4] = (int)(tile_d1 & 0xFFFFu);                      // tile_dim1 (tile_dim2=0)
  g1[5] = (int)(unsigned)(stride0_elems & 0xFFFFFFFFull);
  g1[6] = (int)(unsigned)((stride0_elems >> 32) & 0xFFFFull);
  g1[7] = 0;
  v4i_ z4 = {0, 0, 0, 0};
  v8i_ z8 = {0, 0, 0, 0, 0, 0, 0, 0};
  __builtin_amdgcn_tensor_load_to_lds(g0, g1, z4, z4, z8, 0);
}

// A-fragment (16x32 bf16, ISA 7.12.2): lane holds row M=lane%16;
// elems 0..7 -> K=k0+half*8+e, elems 8..15 -> K=k0+16+half*8+(e-8).
__device__ __forceinline__ v16bf frag_ld_a(const __bf16* base) {
  v8bf lo = *(const v8bf*)base;
  v8bf hi = *(const v8bf*)(base + 16);
  v16bf r;
#pragma unroll
  for (int i = 0; i < 8; ++i) { r[i] = lo[i]; r[i + 8] = hi[i]; }
  return r;
}
__device__ __forceinline__ v8f wmma_bf16(v16bf a, v16bf b, v8f c) {
  return __builtin_amdgcn_wmma_f32_16x16x32_bf16(false, a, false, b,
                                                 (short)0, c, false, false);
}

// ---------------------------------------------------------------------------
// Stage 0 kernels: one-time conversions.
// ---------------------------------------------------------------------------
__global__ __launch_bounds__(256) void convert_x_kernel(
    const float* __restrict__ x, unsigned short* __restrict__ y) {
  int i = (blockIdx.x * 256 + threadIdx.x) * 8;
  float4 a = *(const float4*)(x + i);
  float4 b = *(const float4*)(x + i + 4);
  uint4 o;
  o.x = pack2(a.x, a.y); o.y = pack2(a.z, a.w);
  o.z = pack2(b.x, b.y); o.w = pack2(b.z, b.w);
  *(uint4*)(y + i) = o;
}

// W[K][N] fp32 row-major -> Wt[N][K] bf16 (LDS-tiled 32x32 transpose)
__global__ __launch_bounds__(256) void convert_wt_kernel(
    const float* __restrict__ W, unsigned short* __restrict__ Wt, int K, int N) {
  __shared__ float tile[32][33];
  int k0 = blockIdx.y * 32, n0 = blockIdx.x * 32;
  int tx = threadIdx.x & 31, ty = threadIdx.x >> 5;
#pragma unroll
  for (int i = 0; i < 4; ++i) {
    int r = ty + 8 * i;
    tile[r][tx] = W[(size_t)(k0 + r) * N + n0 + tx];
  }
  __syncthreads();
#pragma unroll
  for (int i = 0; i < 4; ++i) {
    int r = ty + 8 * i;
    Wt[(size_t)(n0 + r) * K + k0 + tx] = bf2u(tile[tx][r]);
  }
}

// ---------------------------------------------------------------------------
// TDM-staged bf16 WMMA GEMM: C[M,N] = A[M,K] @ Bt[N,K]^T
// OUT_MODE: 0 = bf16 row-major, 1 = f32 row-major, 2 = bf16 per-head transposed
// (mode 2: C row = b*S+s, col = h*HD+d  ->  Vt[((b*H+h)*HD+d)*S + s])
// 256 threads = 8 waves (2x4); wave tile 64x32 = 4x2 WMMA accums; BK=32.
// ---------------------------------------------------------------------------
template <int OUT_MODE>
__global__ __launch_bounds__(256) void gemm_bf16t_kernel(
    const unsigned short* __restrict__ A, const unsigned short* __restrict__ Bt,
    void* __restrict__ Cp, int M, int N, int K) {
  __shared__ __align__(32) __bf16 ldsA[2][128 * 32];   // [row][k]
  __shared__ __align__(32) __bf16 ldsB[2][128 * 32];   // [n][k]

  const int t = threadIdx.x;
  const int lane = t & 31, wave = t >> 5;
  const int half = lane >> 4, l16 = lane & 15;
  const int wm = wave >> 2, wn = wave & 3;
  const int m0 = blockIdx.y * 128, n0 = blockIdx.x * 128;

  v8f acc[4][2] = {};

  if (t < 32) {   // wave 0 drives the TDM
    tdm_load_2d(lds_off(&ldsA[0][0]), A + (size_t)m0 * K, 32, 128, K, M, K);
    tdm_load_2d(lds_off(&ldsB[0][0]), Bt + (size_t)n0 * K, 32, 128, K, N, K);
  }

  int cur = 0;
  for (int kt = 0; kt < K; kt += 32) {
    if (t < 32) __builtin_amdgcn_s_wait_tensorcnt(0);
    __syncthreads();                              // buf[cur] ready for everyone
    if (t < 32 && kt + 32 < K) {                  // prefetch next into freed buffer
      tdm_load_2d(lds_off(&ldsA[cur ^ 1][0]), A + (size_t)m0 * K + kt + 32,
                  32, 128, K, M, K);
      tdm_load_2d(lds_off(&ldsB[cur ^ 1][0]), Bt + (size_t)n0 * K + kt + 32,
                  32, 128, K, N, K);
    }
    const __bf16* la = ldsA[cur];
    const __bf16* lb = ldsB[cur];
    v16bf afr[4], bfr[2];
#pragma unroll
    for (int mi = 0; mi < 4; ++mi)
      afr[mi] = frag_ld_a(&la[(wm * 64 + mi * 16 + l16) * 32 + half * 8]);
#pragma unroll
    for (int ni = 0; ni < 2; ++ni)
      bfr[ni] = frag_ld_a(&lb[(wn * 32 + ni * 16 + l16) * 32 + half * 8]);
#pragma unroll
    for (int mi = 0; mi < 4; ++mi)
#pragma unroll
      for (int ni = 0; ni < 2; ++ni)
        acc[mi][ni] = wmma_bf16(afr[mi], bfr[ni], acc[mi][ni]);
    cur ^= 1;
  }

  // epilogue: D layout -> lane holds n = l16, rows j + 8*half
#pragma unroll
  for (int mi = 0; mi < 4; ++mi)
#pragma unroll
    for (int ni = 0; ni < 2; ++ni)
#pragma unroll
      for (int j = 0; j < 8; ++j) {
        int row = m0 + wm * 64 + mi * 16 + j + 8 * half;
        int col = n0 + wn * 32 + ni * 16 + l16;
        float v = acc[mi][ni][j];
        if (OUT_MODE == 1) {
          ((float*)Cp)[(size_t)row * N + col] = v;
        } else if (OUT_MODE == 0) {
          ((unsigned short*)Cp)[(size_t)row * N + col] = bf2u(v);
        } else {  // per-head transposed V: contiguous in seq per lane
          int bb = row >> 11, ss = row & (S_ - 1);
          int hh = col >> 7, dd = col & (HD_ - 1);
          ((unsigned short*)Cp)[(((size_t)bb * H_ + hh) * HD_ + dd) * S_ + ss] =
              bf2u(v);
        }
      }
}

// ---------------------------------------------------------------------------
// Flash attention: block = (128-q tile, h, b); 8 waves x 16 q rows.
// K tile [64 key][128 d] and Vt tile [128 d][64 key] staged by TDM, dbl-buffered.
// ---------------------------------------------------------------------------
__global__ __launch_bounds__(256) void attn_kernel(
    const unsigned short* __restrict__ Qb, const unsigned short* __restrict__ Kb,
    const unsigned short* __restrict__ Vt, unsigned short* __restrict__ Ob) {
  __shared__ __align__(32) __bf16 ldsK[2][64 * 128];    // [key][d]
  __shared__ __align__(32) __bf16 ldsV[2][128 * 64];    // [d][key]
  __shared__ __align__(32) __bf16 ldsP[8 * 16 * 64];    // per-wave P [q][key]

  const int t = threadIdx.x;
  const int lane = t & 31, wave = t >> 5;
  const int half = lane >> 4, l16 = lane & 15;
  const int qt = blockIdx.x, h = blockIdx.y, b = blockIdx.z;

  const int qrow0 = qt * 128 + wave * 16;
  const size_t seqbase = (size_t)b * S_ * D_;
  const int hoff = h * HD_;
  const unsigned short* Kp = Kb + seqbase + hoff;                     // + key*D
  const unsigned short* Vp = Vt + ((size_t)b * H_ + h) * HD_ * S_;    // + d*S + key

  // resident Q fragments (A layout), 4 chunks of 32 along HD
  v16bf qf[4];
#pragma unroll
  for (int dc = 0; dc < 4; ++dc) {
    const __bf16* base = (const __bf16*)Qb + seqbase +
                         (size_t)(qrow0 + l16) * D_ + hoff + dc * 32 + half * 8;
    qf[dc] = frag_ld_a(base);
  }

  v8f o[8] = {};
  float mrow[8], lrow[8];
#pragma unroll
  for (int j = 0; j < 8; ++j) { mrow[j] = -1e30f; lrow[j] = 0.f; }
  const float scale = 0.08838834764831845f;  // 1/sqrt(128)

  if (t < 32) {
    tdm_load_2d(lds_off(&ldsK[0][0]), Kp, 128, 64, D_, S_, D_);
    tdm_load_2d(lds_off(&ldsV[0][0]), Vp, 64, 128, S_, HD_, S_);
  }

  int cur = 0;
  for (int kt = 0; kt < S_ / 64; ++kt) {
    if (t < 32) __builtin_amdgcn_s_wait_tensorcnt(0);
    __syncthreads();
    if (t < 32 && kt + 1 < S_ / 64) {
      tdm_load_2d(lds_off(&ldsK[cur ^ 1][0]), Kp + (size_t)(kt + 1) * 64 * D_,
                  128, 64, D_, S_, D_);
      tdm_load_2d(lds_off(&ldsV[cur ^ 1][0]), Vp + (kt + 1) * 64,
                  64, 128, S_, HD_, S_);
    }
    const __bf16* lk = ldsK[cur];
    const __bf16* lv = ldsV[cur];

    // S = Q @ K^T  (4 key-tiles x 4 d-chunks)
    v8f s[4] = {};
#pragma unroll
    for (int nt = 0; nt < 4; ++nt)
#pragma unroll
      for (int dc = 0; dc < 4; ++dc) {
        v16bf kf = frag_ld_a(&lk[(nt * 16 + l16) * 128 + dc * 32 + half * 8]);
        s[nt] = wmma_bf16(qf[dc], kf, s[nt]);
      }

    // online softmax (row = j + 8*half; 16 lanes share a row)
#pragma unroll
    for (int j = 0; j < 8; ++j) {
      float mc = -1e30f;
#pragma unroll
      for (int nt = 0; nt < 4; ++nt) { s[nt][j] *= scale; mc = fmaxf(mc, s[nt][j]); }
#pragma unroll
      for (int off = 1; off < 16; off <<= 1) mc = fmaxf(mc, __shfl_xor(mc, off, 32));
      float mn = fmaxf(mrow[j], mc);
      float alpha = __expf(mrow[j] - mn);
      mrow[j] = mn;
      float rs = 0.f;
#pragma unroll
      for (int nt = 0; nt < 4; ++nt) {
        float p = __expf(s[nt][j] - mn);
        s[nt][j] = p; rs += p;
      }
#pragma unroll
      for (int off = 1; off < 16; off <<= 1) rs += __shfl_xor(rs, off, 32);
      lrow[j] = lrow[j] * alpha + rs;
#pragma unroll
      for (int ot = 0; ot < 8; ++ot) o[ot][j] *= alpha;
#pragma unroll
      for (int nt = 0; nt < 4; ++nt)
        ldsP[wave * 1024 + (j + 8 * half) * 64 + nt * 16 + l16] = f2bf(s[nt][j]);
    }
    asm volatile("s_wait_dscnt 0x0" ::: "memory");  // wave-private P round-trip

    // O += P @ V  (2 key-chunks x 8 d-tiles)
#pragma unroll
    for (int kc = 0; kc < 2; ++kc) {
      v16bf pf = frag_ld_a(&ldsP[wave * 1024 + l16 * 64 + kc * 32 + half * 8]);
#pragma unroll
      for (int ot = 0; ot < 8; ++ot) {
        v16bf vf = frag_ld_a(&lv[(ot * 16 + l16) * 64 + kc * 32 + half * 8]);
        o[ot] = wmma_bf16(pf, vf, o[ot]);
      }
    }
    cur ^= 1;
  }

  // finalize: O /= l, store bf16 [B,S,D]
#pragma unroll
  for (int ot = 0; ot < 8; ++ot)
#pragma unroll
    for (int j = 0; j < 8; ++j) {
      float v = o[ot][j] / lrow[j];
      int row = qrow0 + j + 8 * half;
      int col = hoff + ot * 16 + l16;
      Ob[seqbase + (size_t)row * D_ + col] = bf2u(v);
    }
}

// ---------------------------------------------------------------------------

extern "C" void kernel_launch(void* const* d_in, const int* in_sizes, int n_in,
                              void* d_out, int out_size, void* d_ws, size_t ws_size,
                              hipStream_t stream) {
  const float* X  = (const float*)d_in[0];
  const float* Wq = (const float*)d_in[1];
  const float* Wk = (const float*)d_in[2];
  const float* Wv = (const float*)d_in[3];
  const float* Wo = (const float*)d_in[4];

  const size_t xe = (size_t)B_ * S_ * D_;  // 8M
  const size_t we = (size_t)D_ * D_;       // 4M
  unsigned short* Xb  = (unsigned short*)d_ws;       // 16 MB
  unsigned short* Wqt = Xb + xe;                     // 8 MB each
  unsigned short* Wkt = Wqt + we;
  unsigned short* Wvt = Wkt + we;
  unsigned short* Wot = Wvt + we;
  unsigned short* Qb  = Wot + we;                    // 16 MB each
  unsigned short* Kb  = Qb + xe;
  unsigned short* Vtb = Kb + xe;
  unsigned short* Ob  = Vtb + xe;                    // total 112 MB

  const int M = B_ * S_;  // 4096
  dim3 blk(256);

  convert_x_kernel<<<(int)(xe / (256 * 8)), blk, 0, stream>>>(X, Xb);
  dim3 gW(D_ / 32, D_ / 32);
  convert_wt_kernel<<<gW, blk, 0, stream>>>(Wq, Wqt, D_, D_);
  convert_wt_kernel<<<gW, blk, 0, stream>>>(Wk, Wkt, D_, D_);
  convert_wt_kernel<<<gW, blk, 0, stream>>>(Wv, Wvt, D_, D_);
  convert_wt_kernel<<<gW, blk, 0, stream>>>(Wo, Wot, D_, D_);

  dim3 gGemm(D_ / 128, M / 128);  // (16, 32)
  gemm_bf16t_kernel<0><<<gGemm, blk, 0, stream>>>(Xb, Wqt, Qb, M, D_, D_);
  gemm_bf16t_kernel<0><<<gGemm, blk, 0, stream>>>(Xb, Wkt, Kb, M, D_, D_);
  gemm_bf16t_kernel<2><<<gGemm, blk, 0, stream>>>(Xb, Wvt, Vtb, M, D_, D_);

  dim3 gAttn(S_ / 128, H_, B_);   // (16, 16, 2)
  attn_kernel<<<gAttn, blk, 0, stream>>>(Qb, Kb, Vtb, Ob);

  gemm_bf16t_kernel<1><<<gGemm, blk, 0, stream>>>(Ob, Wot, (float*)d_out, M, D_, D_);
}
